// Attention_20272245637815
// MI455X (gfx1250) — compile-verified, matrix operand order checked by reading, not run on hardware
//
#include <hip/hip_runtime.h>
#include <hip/hip_bf16.h>
#include <math.h>

// Problem constants (B,N,DIM,H,DH = 2,2048,1024,16,64)
#define B_     2
#define N_     2048
#define DIM_   1024
#define H_     16
#define DH_    64
#define INNER_ 1024
#define M_     (B_*N_)       // 4096 tokens
#define NQKV_  (3*INNER_)    // 3072

typedef __attribute__((ext_vector_type(16))) _Float16 v16h;
typedef __attribute__((ext_vector_type(8)))  float    v8f;
typedef __attribute__((ext_vector_type(4)))  unsigned int u32x4;
typedef __attribute__((ext_vector_type(8)))  int      i32x8;
typedef __attribute__((ext_vector_type(4)))  int      i32x4;

union AFrag { v16h v; _Float16 h[16]; uint4 u[2]; };
union CAcc  { v8f  v; float    f[8];  };

#define WMMA(a,b,c) __builtin_amdgcn_wmma_f32_16x16x32_f16(false,(a),false,(b),(short)0,(c),false,false)

// Load a 16x32 f16 fragment from a row-major matrix (leading dim `ld` elems).
// Per CDNA5 ISA 16-bit A layout: lane L holds row L%16, halves = K {0..7,16..23} + 8*(L>=16).
__device__ __forceinline__ v16h load_frag(const _Float16* base, int ld) {
  const int lane = threadIdx.x & 31;
  const _Float16* p = base + (lane & 15) * ld + ((lane >> 4) << 3);
  AFrag f;
  f.u[0] = *reinterpret_cast<const uint4*>(p);
  f.u[1] = *reinterpret_cast<const uint4*>(p + 16);
  return f.v;
}

// ---- Tensor Data Mover: 2D tile (tile_d1 rows x tile_d0 elems of 2 bytes) -> LDS -------------
// Descriptor per CDNA5 ISA. All inputs must be wave-uniform.
__device__ __forceinline__ void tdm_load_2d(unsigned int lds_off, const void* gaddr,
                                            unsigned int tensor_d0, unsigned int tensor_d1,
                                            unsigned int tile_d0, unsigned int tile_d1,
                                            unsigned long long stride0_elems) {
  const unsigned long long ga = (unsigned long long)gaddr;
  u32x4 g0;
  g0.x = 1u;                                               // count=1, user descriptor
  g0.y = lds_off;                                          // lds_addr (bytes)
  g0.z = (unsigned int)ga;                                 // global_addr[31:0]
  g0.w = (unsigned int)((ga >> 32) & 0x1FFFFFFull) | (2u << 30);  // global_addr[56:32] | type=2
  i32x8 g1;
  g1[0] = (int)(1u << 16);                                 // data_size=1 (2 bytes)
  g1[1] = (int)((tensor_d0 & 0xFFFFu) << 16);              // tensor_dim0[15:0] at bits 63:48
  g1[2] = (int)(((tensor_d0 >> 16) & 0xFFFFu) | ((tensor_d1 & 0xFFFFu) << 16));
  g1[3] = (int)(((tensor_d1 >> 16) & 0xFFFFu) | ((tile_d0 & 0xFFFFu) << 16));  // tile_dim0
  g1[4] = (int)(tile_d1 & 0xFFFFu);                        // tile_dim1 (tile_dim2=0)
  g1[5] = (int)(unsigned int)(stride0_elems & 0xFFFFFFFFull);
  g1[6] = (int)(unsigned int)((stride0_elems >> 32) & 0xFFFFull);  // dim1_stride unused (2D)
  g1[7] = 0;
  i32x4 z4 = {0, 0, 0, 0};
#if __clang_major__ >= 23
  i32x8 z8 = {0, 0, 0, 0, 0, 0, 0, 0};
  __builtin_amdgcn_tensor_load_to_lds(g0, g1, z4, z4, z8, 0);
#else
  __builtin_amdgcn_tensor_load_to_lds(g0, g1, z4, z4, 0);
#endif
}

// ---------------- conversion kernels ----------------
__global__ void cvt_f32_to_f16(const float* __restrict__ in, _Float16* __restrict__ out, int n) {
  int i = blockIdx.x * 256 + threadIdx.x;
  if (i < n) out[i] = (_Float16)in[i];
}

// in[rows,cols] f32 -> out[cols,rows] f16   (weights stored as [K,Nout] -> [Nout,K])
__global__ void cvt_transpose_f16(const float* __restrict__ in, _Float16* __restrict__ out,
                                  int rows, int cols) {
  int i = blockIdx.x * 256 + threadIdx.x;
  if (i < rows * cols) {
    int r = i / cols, c = i - r * cols;
    out[(size_t)c * rows + r] = (_Float16)in[i];
  }
}

// ---------------- QKV GEMM:  C[4096,3072] = x16[4096,1024] * wqkvT[3072,1024]^T ----------------
// epilogue scatters into q[B,H,N,DH], k[B,H,N,DH], vT[B,H,DH,N] (f16)
__global__ __launch_bounds__(256) void qkv_gemm(const _Float16* __restrict__ A,
                                                const _Float16* __restrict__ Bm,
                                                _Float16* __restrict__ qb,
                                                _Float16* __restrict__ kb,
                                                _Float16* __restrict__ vT) {
  const int wave = threadIdx.x >> 5, lane = threadIdx.x & 31;
  const int m0 = blockIdx.x * 128 + (wave & 1) * 64;   // 4 M-subtiles
  const int n0 = blockIdx.y * 128 + (wave >> 1) * 32;  // 2 N-subtiles
  v8f acc[4][2];
  { v8f z = {};
#pragma unroll
    for (int i = 0; i < 4; ++i) for (int j = 0; j < 2; ++j) acc[i][j] = z; }

  for (int k = 0; k < DIM_; k += 32) {
    v16h a[4], b[2];
#pragma unroll
    for (int i = 0; i < 4; ++i) a[i] = load_frag(A + (size_t)(m0 + 16*i) * DIM_ + k, DIM_);
#pragma unroll
    for (int j = 0; j < 2; ++j) b[j] = load_frag(Bm + (size_t)(n0 + 16*j) * DIM_ + k, DIM_);
#pragma unroll
    for (int i = 0; i < 4; ++i)
#pragma unroll
      for (int j = 0; j < 2; ++j) acc[i][j] = WMMA(a[i], b[j], acc[i][j]);
  }

  const int hi = lane >> 4, lo = lane & 15;
#pragma unroll
  for (int j = 0; j < 2; ++j) {
    // Tile-base column is wave-uniform; a 16-wide tile never straddles part/head boundaries.
    const int cb   = __builtin_amdgcn_readfirstlane(n0 + 16*j);
    const int part = cb >> 10;          // 0:q 1:k 2:v  (scalar branch)
    const int head = (cb & 1023) >> 6;
    const int d    = (cb & 63) + lo;
#pragma unroll
    for (int i = 0; i < 4; ++i) {
      const int mb = __builtin_amdgcn_readfirstlane(m0 + 16*i);  // uniform tile row base
      const int bb = mb >> 11;
      const int nb = mb & (N_ - 1);
      const size_t bh = (size_t)(bb * H_ + head);
      CAcc cc; cc.v = acc[i][j];
      if (part == 0) {
        _Float16* dst = qb + (bh * N_ + nb) * DH_ + d;
#pragma unroll
        for (int r = 0; r < 8; ++r) dst[(size_t)(r + 8*hi) * DH_] = (_Float16)cc.f[r];
      } else if (part == 1) {
        _Float16* dst = kb + (bh * N_ + nb) * DH_ + d;
#pragma unroll
        for (int r = 0; r < 8; ++r) dst[(size_t)(r + 8*hi) * DH_] = (_Float16)cc.f[r];
      } else {
        _Float16* dst = vT + (bh * DH_ + d) * N_ + nb;
#pragma unroll
        for (int r = 0; r < 8; ++r) dst[r + 8*hi] = (_Float16)cc.f[r];
      }
    }
  }
}

// ---------------- L2 normalize q,k rows; fold exp(min(logit_scale, ln100)) into q --------------
__global__ void normalize_qk(_Float16* __restrict__ qb, _Float16* __restrict__ kb,
                             const float* __restrict__ logit_scale) {
  const int ROWS = B_ * H_ * N_;                   // 65536 rows per buffer
  int tid = blockIdx.x * 256 + threadIdx.x;
  if (tid >= 2 * ROWS) return;
  const int which = (tid >= ROWS);
  const int idx   = which ? tid - ROWS : tid;
  const int h     = (idx >> 11) & (H_ - 1);
  _Float16* row = (which ? kb : qb) + (size_t)idx * DH_;

  uint4 d[8];
  const uint4* rp = reinterpret_cast<const uint4*>(row);
#pragma unroll
  for (int i = 0; i < 8; ++i) d[i] = rp[i];
  _Float16* hp = reinterpret_cast<_Float16*>(d);
  float ss = 0.f;
#pragma unroll
  for (int i = 0; i < 64; ++i) { float v = (float)hp[i]; ss += v * v; }
  float s = 1.0f / fmaxf(sqrtf(ss), 1e-12f);
  if (!which) s *= __expf(fminf(logit_scale[h], 4.60517019f));  // ln(1/0.01)
#pragma unroll
  for (int i = 0; i < 64; ++i) hp[i] = (_Float16)((float)hp[i] * s);
  uint4* wp = reinterpret_cast<uint4*>(row);
#pragma unroll
  for (int i = 0; i < 8; ++i) wp[i] = d[i];
}

// ---------------- flash attention: per wave 16 queries, stream 128-key blocks ------------------
// K/V blocks are DMA'd into double-buffered LDS by the Tensor Data Mover: the leader wave issues
// the DMA for block i+1 right after the barrier publishing block i, so the TDM fills one buffer
// while all 8 waves compute from the other. TENSORcnt ops complete in order, so
// s_wait_tensorcnt(0) at the top of iteration i drains exactly block i's two loads.
// S^T = K*Q^T so the f32 C-layout of S^T is directly reusable as the f16 A-fragment layout for P.
__global__ __launch_bounds__(256) void flash_attn(const _Float16* __restrict__ qbuf,
                                                  const _Float16* __restrict__ kbuf,
                                                  const _Float16* __restrict__ vT,
                                                  _Float16* __restrict__ ao) {
  __shared__ _Float16 ldsK[2][128 * DH_];   // [buf][128 keys][64 d]   2 x 16 KB
  __shared__ _Float16 ldsV[2][DH_ * 128];   // [buf][64 d][128 keys]   2 x 16 KB

  const int wave = threadIdx.x >> 5, lane = threadIdx.x & 31, hi = lane >> 4, lo = lane & 15;
  const int bh = blockIdx.y;                    // b*H + h
  const int b  = bh >> 4, h = bh & 15;
  const int q0 = blockIdx.x * 128 + wave * 16;
  const bool leader = (__builtin_amdgcn_readfirstlane((int)wave) == 0);

  const _Float16* qp = qbuf + ((size_t)bh * N_ + q0) * DH_;
  const _Float16* kp = kbuf + (size_t)bh * N_ * DH_;
  const _Float16* vp = vT   + (size_t)bh * DH_ * N_;

  unsigned int offK[2], offV[2];
#pragma unroll
  for (int c = 0; c < 2; ++c) {
    offK[c] = (unsigned int)(uintptr_t)(&ldsK[c][0]);
    offV[c] = (unsigned int)(uintptr_t)(&ldsV[c][0]);
  }

  const v16h bq0 = load_frag(qp,      DH_);     // Q as B operand (queries = columns)
  const v16h bq1 = load_frag(qp + 32, DH_);

  float mrun = -1e30f, lrun = 0.f;
  v8f oacc[4];
  { v8f z = {}; for (int i = 0; i < 4; ++i) oacc[i] = z; }

  const int NBLK = N_ / 128;                    // 16 key blocks
  if (leader) {
    // preload block 0 into buffer 0
    tdm_load_2d(offK[0], kp, DH_, N_, DH_, 128, DH_);
    tdm_load_2d(offV[0], vp, N_, DH_, 128, DH_, N_);
  }

  for (int ib = 0; ib < NBLK; ++ib) {
    const int cur = ib & 1;
    const int kb0 = ib * 128;
    if (leader) __builtin_amdgcn_s_wait_tensorcnt(0);   // block ib's tiles resident
    __syncthreads();                                    // publish buf[cur]; prior reads of buf[cur^1] retired
    if (leader && (ib + 1) < NBLK) {                    // async prefetch of block ib+1
      const int kb1 = kb0 + 128;
      tdm_load_2d(offK[cur ^ 1], kp + (size_t)kb1 * DH_, DH_, N_, DH_, 128, DH_);
      tdm_load_2d(offV[cur ^ 1], vp + kb1, N_, DH_, 128, DH_, N_);
    }

    CAcc s[8];
#pragma unroll
    for (int j = 0; j < 8; ++j) {               // S^T tiles: 16 keys x 16 queries
      const _Float16* pk = &ldsK[cur][(16*j + lo) * DH_ + (hi << 3)];
      AFrag ak0, ak1;
      ak0.u[0] = *reinterpret_cast<const uint4*>(pk);
      ak0.u[1] = *reinterpret_cast<const uint4*>(pk + 16);
      ak1.u[0] = *reinterpret_cast<const uint4*>(pk + 32);
      ak1.u[1] = *reinterpret_cast<const uint4*>(pk + 48);
      v8f z = {};
      s[j].v = WMMA(ak0.v, bq0, z);
      s[j].v = WMMA(ak1.v, bq1, s[j].v);
    }
    // online softmax (per-lane state is per-query; halves combined via xor-16 shuffle)
    float mx = -1e30f;
#pragma unroll
    for (int j = 0; j < 8; ++j)
#pragma unroll
      for (int r = 0; r < 8; ++r) mx = fmaxf(mx, s[j].f[r]);
    mx = fmaxf(mx, __shfl_xor(mx, 16, 32));
    const float mnew = fmaxf(mrun, mx);
    const float corr = __expf(mrun - mnew);
    float rs = 0.f;
    AFrag ap[4];
#pragma unroll
    for (int t = 0; t < 4; ++t) {
#pragma unroll
      for (int r = 0; r < 8; ++r) { float p = __expf(s[2*t  ].f[r] - mnew); rs += p; ap[t].h[r]     = (_Float16)p; }
#pragma unroll
      for (int r = 0; r < 8; ++r) { float p = __expf(s[2*t+1].f[r] - mnew); rs += p; ap[t].h[r + 8] = (_Float16)p; }
    }
    rs += __shfl_xor(rs, 16, 32);
    lrun = lrun * corr + rs;
    mrun = mnew;
    // rescale O: row of O-tile VGPR r is query r+8*hi -> fetch its corr from lane (r+8*hi)
    float cr[8];
#pragma unroll
    for (int r = 0; r < 8; ++r) cr[r] = __shfl(corr, r + 8*hi, 32);
#pragma unroll
    for (int ds = 0; ds < 4; ++ds) {
      CAcc o; o.v = oacc[ds];
#pragma unroll
      for (int r = 0; r < 8; ++r) o.f[r] *= cr[r];
      oacc[ds] = o.v;
    }
    // O += P * V   (V^T tile in LDS: [64 d][128 keys] -> natural B fragments)
#pragma unroll
    for (int ds = 0; ds < 4; ++ds)
#pragma unroll
      for (int t = 0; t < 4; ++t) {
        const _Float16* pv = &ldsV[cur][(ds*16 + lo) * 128 + 32*t + (hi << 3)];
        AFrag bv;
        bv.u[0] = *reinterpret_cast<const uint4*>(pv);
        bv.u[1] = *reinterpret_cast<const uint4*>(pv + 16);
        oacc[ds] = WMMA(ap[t].v, bv.v, oacc[ds]);
      }
  }
  // final 1/l scale + store to attn-out [b, tok, h*64+d] (f16)
  float linv[8];
#pragma unroll
  for (int r = 0; r < 8; ++r) linv[r] = 1.0f / __shfl(lrun, r + 8*hi, 32);
#pragma unroll
  for (int ds = 0; ds < 4; ++ds) {
    CAcc o; o.v = oacc[ds];
    const int c = h * DH_ + ds * 16 + lo;
#pragma unroll
    for (int r = 0; r < 8; ++r) {
      const int tok = q0 + r + 8*hi;
      ao[((size_t)b * N_ + tok) * INNER_ + c] = (_Float16)(o.f[r] * linv[r]);
    }
  }
}

// ---------------- output GEMM: out[4096,1024] = ao[4096,1024] * woutT[1024,1024]^T + bias -------
__global__ __launch_bounds__(256) void out_gemm(const _Float16* __restrict__ A,
                                                const _Float16* __restrict__ Bm,
                                                const float* __restrict__ bias,
                                                float* __restrict__ out) {
  const int wave = threadIdx.x >> 5, lane = threadIdx.x & 31, hi = lane >> 4;
  const int m0 = blockIdx.x * 128 + (wave & 1) * 64;
  const int n0 = blockIdx.y * 128 + (wave >> 1) * 32;
  v8f acc[4][2];
  { v8f z = {}; for (int i = 0; i < 4; ++i) for (int j = 0; j < 2; ++j) acc[i][j] = z; }

  for (int k = 0; k < INNER_; k += 32) {
    v16h a[4], b[2];
#pragma unroll
    for (int i = 0; i < 4; ++i) a[i] = load_frag(A + (size_t)(m0 + 16*i) * INNER_ + k, INNER_);
#pragma unroll
    for (int j = 0; j < 2; ++j) b[j] = load_frag(Bm + (size_t)(n0 + 16*j) * INNER_ + k, INNER_);
#pragma unroll
    for (int i = 0; i < 4; ++i)
#pragma unroll
      for (int j = 0; j < 2; ++j) acc[i][j] = WMMA(a[i], b[j], acc[i][j]);
  }
#pragma unroll
  for (int j = 0; j < 2; ++j) {
    const int c = n0 + 16*j + (lane & 15);
    const float bc = bias[c];
#pragma unroll
    for (int i = 0; i < 4; ++i) {
      CAcc cc; cc.v = acc[i][j];
#pragma unroll
      for (int r = 0; r < 8; ++r) {
        const int m = m0 + 16*i + r + 8*hi;
        out[(size_t)m * DIM_ + c] = cc.f[r] + bc;
      }
    }
  }
}

// ---------------- host launcher ----------------
extern "C" void kernel_launch(void* const* d_in, const int* in_sizes, int n_in,
                              void* d_out, int out_size, void* d_ws, size_t ws_size,
                              hipStream_t stream) {
  (void)in_sizes; (void)n_in; (void)out_size; (void)ws_size;
  const float* x           = (const float*)d_in[0];
  const float* w_qkv       = (const float*)d_in[1];
  const float* w_out       = (const float*)d_in[2];
  const float* b_out       = (const float*)d_in[3];
  const float* logit_scale = (const float*)d_in[4];
  float* out = (float*)d_out;

  char* ws = (char*)d_ws;
  _Float16* x16   = (_Float16*)ws;  ws += (size_t)M_ * DIM_   * 2;   // 8 MB
  _Float16* wqkvT = (_Float16*)ws;  ws += (size_t)NQKV_ * DIM_ * 2;  // 6 MB
  _Float16* woutT = (_Float16*)ws;  ws += (size_t)DIM_ * INNER_ * 2; // 2 MB
  _Float16* qb    = (_Float16*)ws;  ws += (size_t)B_*H_*N_*DH_ * 2;  // 8 MB
  _Float16* kb    = (_Float16*)ws;  ws += (size_t)B_*H_*N_*DH_ * 2;  // 8 MB
  _Float16* vT    = (_Float16*)ws;  ws += (size_t)B_*H_*N_*DH_ * 2;  // 8 MB
  _Float16* ao    = (_Float16*)ws;                                   // 8 MB

  // 1) precision conversion / weight transposition
  { int n = M_ * DIM_;      cvt_f32_to_f16<<<(n + 255) / 256, 256, 0, stream>>>(x, x16, n); }
  { int n = DIM_ * NQKV_;   cvt_transpose_f16<<<(n + 255) / 256, 256, 0, stream>>>(w_qkv, wqkvT, DIM_, NQKV_); }
  { int n = INNER_ * DIM_;  cvt_transpose_f16<<<(n + 255) / 256, 256, 0, stream>>>(w_out, woutT, INNER_, DIM_); }

  // 2) QKV projection (WMMA)
  qkv_gemm<<<dim3(M_ / 128, NQKV_ / 128), 256, 0, stream>>>(x16, wqkvT, qb, kb, vT);

  // 3) L2 normalize q,k + fold per-head temperature into q
  { int n = 2 * B_ * H_ * N_; normalize_qk<<<(n + 255) / 256, 256, 0, stream>>>(qb, kb, logit_scale); }

  // 4) flash attention (WMMA + double-buffered TDM K/V staging)
  flash_attn<<<dim3(N_ / 128, B_ * H_), 256, 0, stream>>>(qb, kb, vT, ao);

  // 5) output projection + bias (WMMA)
  out_gemm<<<dim3(M_ / 128, DIM_ / 128), 256, 0, stream>>>(ao, woutT, b_out, out);
}
